// OutputLayer_53463752901308
// MI455X (gfx1250) — matrix operand order repaired and checked by reading, not compiled, not run on hardware
//
#include <hip/hip_runtime.h>

// CDNA5 / gfx1250 implementation of the SNN output layer.
//
// Math decomposition: only out[:, b, 255, :] is ever needed for the final
// result; BN mean/var are collapsed into column sums and Gram matrices:
//   mean_b  = (sumTS_b . colsumW + T*S*sum(bias)) / (T*S*V)
//   E[o^2]b = (sum(G_b * C) + 2*sumTS_b.colsumWb + T*S*sum(b^2)) / (T*S*V)
// with C = W^T W (heavy: 19 G MACs over L2-resident f16 W -> WMMA) and
// G_b = S1^T S1 (exact in f16 WMMA: spikes are {0,1}, sums <= 1024 < 2^24).
// The threshold-critical row matmul (8 x 32000 dots) stays exact f32 and is
// HBM-bound on one read of W; its s1-row staging uses the Tensor Data Mover.

typedef _Float16 v4h  __attribute__((ext_vector_type(4)));
typedef _Float16 v8h  __attribute__((ext_vector_type(8)));
typedef _Float16 v16h __attribute__((ext_vector_type(16)));
typedef float    v8f  __attribute__((ext_vector_type(8)));
typedef unsigned int v4u __attribute__((ext_vector_type(4)));
typedef int      v8i  __attribute__((ext_vector_type(8)));
typedef int      v4i  __attribute__((ext_vector_type(4)));

namespace snncfg {
constexpr int T = 4, NB = 2, S = 256, E = 768, V = 32000;
constexpr int KS = 40;                 // LDS k-stride (halves): 80B row => 16B-aligned b128 frag reads
constexpr float TSV = 4.0f * 256.0f * 32000.0f;

constexpr size_t alignup(size_t x) { return (x + 255) & ~size_t(255); }
constexpr size_t WH_OFF    = 0;                                            // W as f16 [V][E]
constexpr size_t S1_OFF    = alignup(WH_OFF + (size_t)V * E * 2);          // spikes f16 [T][NB][S][E]
constexpr size_t ROW_OFF   = alignup(S1_OFF + (size_t)T * NB * S * E * 2); // out_row f32 [T*NB][V]
constexpr size_t C_OFF     = alignup(ROW_OFF + (size_t)T * NB * V * 4);    // C f32 [E][E]   (zeroed)
constexpr size_t G_OFF     = C_OFF + (size_t)E * E * 4;                    // G f32 [NB][E][E]
constexpr size_t CSW_OFF   = G_OFF + (size_t)NB * E * E * 4;               // colsumW f32 [E]
constexpr size_t CSWB_OFF  = CSW_OFF + (size_t)E * 4;                      // colsum(W*b) f32 [E]
constexpr size_t SUMTS_OFF = CSWB_OFF + (size_t)E * 4;                     // sumTS f32 [NB][E]
constexpr size_t E2_OFF    = SUMTS_OFF + (size_t)NB * E * 4;               // Gram part of E[o^2] [NB]
constexpr size_t MSUM_OFF  = E2_OFF + (size_t)NB * 4;                      // sum_v m [NB]
constexpr size_t STATS_OFF = MSUM_OFF + (size_t)NB * 4;                    // (mean, invstd) [NB]
constexpr size_t END_OFF   = STATS_OFF + (size_t)2 * NB * 4;
constexpr int    ZERO_N    = (int)((END_OFF - C_OFF) / 4);
} // namespace snncfg
using namespace snncfg;

__device__ __forceinline__ float snn_block_reduce(float v, float* red) {
  int tid = threadIdx.x;
  red[tid] = v; __syncthreads();
  for (int s = 128; s > 0; s >>= 1) {
    if (tid < s) red[tid] += red[tid + s];
    __syncthreads();
  }
  float r = red[0]; __syncthreads();
  return r;
}

// ---------------------------------------------------------------- zeroing ---
__global__ void snn_zero_kernel(float* p, int n) {
  for (int i = blockIdx.x * blockDim.x + threadIdx.x; i < n; i += gridDim.x * blockDim.x)
    p[i] = 0.0f;
}

// ------------------------------------------- W -> f16, column sums of W -----
__global__ __launch_bounds__(256) void snn_convert_kernel(
    const float* __restrict__ W, _Float16* __restrict__ WH,
    float* __restrict__ colsumW, float* __restrict__ colsumWb,
    const float* __restrict__ bias) {
  int tid = threadIdx.x;
  int row0 = blockIdx.x * 32;
  float a0 = 0, a1 = 0, a2 = 0, w0 = 0, w1 = 0, w2 = 0;
  for (int r = 0; r < 32; ++r) {
    int row = row0 + r;
    float bv = bias[row];
    size_t base = (size_t)row * E;
    float v0 = W[base + tid], v1 = W[base + tid + 256], v2 = W[base + tid + 512];
    WH[base + tid]       = (_Float16)v0;
    WH[base + tid + 256] = (_Float16)v1;
    WH[base + tid + 512] = (_Float16)v2;
    a0 += v0; a1 += v1; a2 += v2;
    w0 += v0 * bv; w1 += v1 * bv; w2 += v2 * bv;
  }
  atomicAdd(&colsumW[tid],        a0);
  atomicAdd(&colsumW[tid + 256],  a1);
  atomicAdd(&colsumW[tid + 512],  a2);
  atomicAdd(&colsumWb[tid],       w0);
  atomicAdd(&colsumWb[tid + 256], w1);
  atomicAdd(&colsumWb[tid + 512], w2);
}

// ------------------------------------------- IF neuron #1 over x -> spikes --
__global__ __launch_bounds__(256) void snn_if1_kernel(
    const float* __restrict__ x, _Float16* __restrict__ s1h,
    float* __restrict__ sumTS) {
  int g = blockIdx.x * 256 + threadIdx.x;     // g = b*E + e, 0..1535
  int b = g / E, e = g % E;
  int s0 = blockIdx.y * 64;
  float cnt = 0.0f;
  for (int s = s0; s < s0 + 64; ++s) {
    float v = 0.0f;
    size_t base = ((size_t)b * S + s) * E + e;
    #pragma unroll
    for (int t = 0; t < T; ++t) {
      size_t idx = (size_t)t * NB * S * E + base;
      v += x[idx];
      bool sp = (v >= 1.0f);
      s1h[idx] = (_Float16)(sp ? 1.0f : 0.0f);
      if (sp) { v = 0.0f; cnt += 1.0f; }
    }
  }
  atomicAdd(&sumTS[g], cnt);
}

// -------------------------- WMMA Gram kernel:  out[b] += X_chunk^T X_chunk --
// REMAP=false: row(r) = r (W case).   REMAP=true (spike tensor [T][NB][S][E]):
//   row(r) = (r>>8)*outerStride + (batch<<8) + (r&255)
template <bool REMAP>
__device__ __forceinline__ int snn_maprow(int r, int batch, int outerStride) {
  if constexpr (REMAP)
    return ((r >> 8) * outerStride) + (batch << 8) + (r & 255);
  else
    return r;
}

template <bool REMAP>
__global__ __launch_bounds__(256) void snn_gram_kernel(
    const _Float16* __restrict__ X, float* __restrict__ Cout,
    int K, int kChunks, int outerStride) {
  const int batch = blockIdx.z / kChunks;
  const int chunk = blockIdx.z % kChunks;
  const int kPer  = K / kChunks;
  const int k0    = chunk * kPer;
  const int tileM = blockIdx.x * 128;
  const int tileN = blockIdx.y * 128;
  float* outBase = Cout + (size_t)batch * E * E;

  // double-buffered, [e][k]-transposed staging tiles (2*2*10240 B = 40 KB)
  __shared__ _Float16 lds[2][2][128 * KS];

  const int tid  = threadIdx.x;
  const int wave = tid >> 5, lane = tid & 31;
  const int half = lane >> 4, l16 = lane & 15;
  const int wr = wave >> 2, wc = wave & 3;   // 2x4 wave grid; each wave: 4 M x 2 N subtiles

  // staging decomposition: 32(k) x 16(uint4) per tile, 2 passes of 256 threads
  const int sk0 = tid >> 4;                  // k row (pass 0); +16 for pass 1
  const int sq  = tid & 15;                  // uint4 column (8 halves)

  v8f acc[4][2];
  #pragma unroll
  for (int i = 0; i < 4; ++i)
    #pragma unroll
    for (int j = 0; j < 2; ++j)
      acc[i][j] = (v8f)0.0f;

  uint4 st[2][2];                            // [pass][A/B] in-flight staging regs
  auto do_load = [&](int kbase) {
    #pragma unroll
    for (int i = 0; i < 2; ++i) {
      int row = snn_maprow<REMAP>(kbase + sk0 + i * 16, batch, outerStride);
      const uint4* src = (const uint4*)(X + (size_t)row * E);
      st[i][0] = src[(tileM >> 3) + sq];
      st[i][1] = src[(tileN >> 3) + sq];
    }
  };
  auto do_store = [&](int buf) {
    #pragma unroll
    for (int i = 0; i < 2; ++i) {
      int k = sk0 + i * 16;
      int e = sq * 8;
      #pragma unroll
      for (int ab = 0; ab < 2; ++ab) {
        const _Float16* h = (const _Float16*)&st[i][ab];
        #pragma unroll
        for (int j = 0; j < 8; ++j)
          lds[buf][ab][(e + j) * KS + k] = h[j];
      }
    }
  };

  const int nIter = kPer >> 5;
  do_load(k0);
  do_store(0);
  __syncthreads();

  for (int it = 0; it < nIter; ++it) {
    const int buf = it & 1;
    const bool more = (it + 1 < nIter);
    if (more) do_load(k0 + (it + 1) * 32);   // overlap global latency with WMMA

    const _Float16* lA = lds[buf][0];
    const _Float16* lB = lds[buf][1];
    // fragments per documented 16-bit A(16x32)/B(32x16) lane layouts
    v16h a[4], b[2];
    #pragma unroll
    for (int i = 0; i < 4; ++i) {
      int e = (wr * 4 + i) * 16 + l16;
      // lanes 0-15: K=0..7 & 16..23 ; lanes 16-31: K=8..15 & 24..31
      v8h lo = *(const v8h*)(lA + e * KS + half * 8);
      v8h hi = *(const v8h*)(lA + e * KS + half * 8 + 16);
      a[i] = __builtin_shufflevector(lo, hi, 0,1,2,3,4,5,6,7,8,9,10,11,12,13,14,15);
    }
    #pragma unroll
    for (int j = 0; j < 2; ++j) {
      int e = (wc * 2 + j) * 16 + l16;
      // lanes 0-15: K=0..7 & 16..23 ; lanes 16-31: K=8..15 & 24..31
      v8h lo = *(const v8h*)(lB + e * KS + half * 8);
      v8h hi = *(const v8h*)(lB + e * KS + half * 8 + 16);
      b[j] = __builtin_shufflevector(lo, hi, 0,1,2,3,4,5,6,7,8,9,10,11,12,13,14,15);
    }
    #pragma unroll
    for (int i = 0; i < 4; ++i)
      #pragma unroll
      for (int j = 0; j < 2; ++j)
        acc[i][j] = __builtin_amdgcn_wmma_f32_16x16x32_f16(
            false, a[i], false, b[j], (short)0, acc[i][j], false, false);

    if (more) do_store(buf ^ 1);
    __syncthreads();
  }

  // ---- K-split partials: f32 atomic reduction into the (zeroed) output.
  #pragma unroll
  for (int i = 0; i < 4; ++i) {
    int rowBase = tileM + (wr * 4 + i) * 16 + half * 8;   // C/D: VGPR p -> M = p + half*8
    #pragma unroll
    for (int j = 0; j < 2; ++j) {
      int col = tileN + (wc * 2 + j) * 16 + l16;
      #pragma unroll
      for (int p = 0; p < 8; ++p)
        atomicAdd(&outBase[(size_t)(rowBase + p) * E + col], acc[i][j][p]);
    }
  }
}

// ---------------------------------- sum(G_b .* C) -> E2 partial (Gram part) -
__global__ __launch_bounds__(256) void snn_reduceE2_kernel(
    const float* __restrict__ G, const float* __restrict__ C, float* __restrict__ E2) {
  __shared__ float red[256];
  int b = blockIdx.y;
  size_t base = (size_t)blockIdx.x * 2048;
  const float* Gb = G + (size_t)b * E * E;
  float p = 0.0f;
  #pragma unroll
  for (int i = 0; i < 8; ++i) {
    size_t idx = base + i * 256 + threadIdx.x;
    p += Gb[idx] * C[idx];
  }
  float s = snn_block_reduce(p, red);
  if (threadIdx.x == 0) atomicAdd(&E2[b], s);
}

// ----------------------------- exact f32 row matmul: out_row = s1_row . W^T -
// s1 row slice (8 x 768 f16, row stride S*E) is staged to LDS by the Tensor
// Data Mover: D# built per CDNA5 ISA, issued by wave 0,
// drained with s_wait_tensorcnt, published by the workgroup barrier.
// This toolchain exposes the 6-arg builtin: (g0 v4u, g1 v8i, g2 v4i, g3 v4i,
// g4 v8i, cpol) — groups beyond 0/1 are zero/disabled for a 2D tile.
__global__ __launch_bounds__(256) void snn_rowmat_kernel(
    const float* __restrict__ W, const _Float16* __restrict__ s1h,
    const float* __restrict__ bias, float* __restrict__ out_row) {
  __shared__ _Float16 srow[T * NB][E];       // 12 KB
  int tid = threadIdx.x;
  if (tid < 32) {                            // one wave issues the TDM op
#if __has_builtin(__builtin_amdgcn_tensor_load_to_lds)
    unsigned int lds_addr = (unsigned int)(uintptr_t)(&srow[0][0]);
    unsigned long long ga =
        (unsigned long long)(uintptr_t)(s1h + (size_t)(S - 1) * E);
    // D# group 0: count=1, lds_addr, global_addr[56:0], type=2
    v4u g0 = { 1u, lds_addr, (unsigned int)ga,
               ((unsigned int)(ga >> 32) & 0x01ffffffu) | (2u << 30) };
    // D# group 1: data_size=2B; tensor_dim0=E, tensor_dim1=8; tile 768x8;
    //             tensor_dim0_stride = S*E (halves between consecutive tb rows)
    v8i g1 = { (int)(1u << 16),
               (int)((unsigned)E << 16),
               (int)(8u << 16),
               (int)((unsigned)E << 16),
               (int)(T * NB),
               (int)(S * E),
               0, 0 };
    v4i g2 = { 0, 0, 0, 0 };
    v4i g3 = { 0, 0, 0, 0 };
    v8i g4 = { 0, 0, 0, 0, 0, 0, 0, 0 };
    __builtin_amdgcn_tensor_load_to_lds(g0, g1, g2, g3, g4, 0);
    __builtin_amdgcn_s_wait_tensorcnt(0);
#else
    for (int i = tid; i < T * NB * E; i += 32) {
      int tb = i / E, e = i % E;
      srow[tb][e] = s1h[((size_t)tb * S + (S - 1)) * E + e];
    }
#endif
  }
  __syncthreads();

  int wave = tid >> 5, lane = tid & 31;
  int v = blockIdx.x * 8 + wave;
  const float4* W4 = (const float4*)(W + (size_t)v * E);
  float acc[T * NB];
  #pragma unroll
  for (int tb = 0; tb < T * NB; ++tb) acc[tb] = 0.0f;
  #pragma unroll
  for (int i = 0; i < E / (32 * 4); ++i) {
    int c4 = i * 32 + lane;
    float4 w = W4[c4];
    #pragma unroll
    for (int tb = 0; tb < T * NB; ++tb) {
      v4h s = *(const v4h*)&srow[tb][c4 * 4];
      acc[tb] += w.x * (float)s[0] + w.y * (float)s[1] +
                 w.z * (float)s[2] + w.w * (float)s[3];
    }
  }
  #pragma unroll
  for (int tb = 0; tb < T * NB; ++tb)
    for (int off = 16; off >= 1; off >>= 1)
      acc[tb] += __shfl_down(acc[tb], off, 32);
  if (lane == 0) {
    float bv = bias[v];
    #pragma unroll
    for (int tb = 0; tb < T * NB; ++tb)
      out_row[(size_t)tb * V + v] = acc[tb] + bv;
  }
}

// ----------------------------------------------------- BN statistics --------
__global__ __launch_bounds__(256) void snn_meanvar_kernel(
    const float* __restrict__ sumTS, const float* __restrict__ colsumW,
    const float* __restrict__ colsumWb, const float* __restrict__ bias,
    const float* __restrict__ E2, float* __restrict__ stats) {
  __shared__ float red[256];
  int tid = threadIdx.x;
  float pb = 0, pb2 = 0;
  for (int v = tid; v < V; v += 256) { float bv = bias[v]; pb += bv; pb2 += bv * bv; }
  float sumB  = snn_block_reduce(pb,  red);
  float sumB2 = snn_block_reduce(pb2, red);
  for (int b = 0; b < NB; ++b) {
    float pw = 0, pwb = 0;
    for (int e = tid; e < E; e += 256) {
      float st = sumTS[b * E + e];
      pw  += st * colsumW[e];
      pwb += st * colsumWb[e];
    }
    float dotW  = snn_block_reduce(pw,  red);
    float dotWb = snn_block_reduce(pwb, red);
    if (tid == 0) {
      float mean = (dotW + (float)(T * S) * sumB) / TSV;
      float e2   = (E2[b] + 2.0f * dotWb + (float)(T * S) * sumB2) / TSV;
      float var  = e2 - mean * mean;
      stats[2 * b]     = mean;
      stats[2 * b + 1] = rsqrtf(var + 1e-5f);
    }
    __syncthreads();
  }
}

// ------------------------------ BN + IF neuron #2 + time-mean (unnormalized)-
__global__ __launch_bounds__(256) void snn_finalize1_kernel(
    const float* __restrict__ out_row, const float* __restrict__ stats,
    float* __restrict__ out, float* __restrict__ msum) {
  __shared__ float red[256];
  int b = blockIdx.x / (V / 256);
  int v = (blockIdx.x % (V / 256)) * 256 + threadIdx.x;
  float mean = stats[2 * b], inv = stats[2 * b + 1];
  float vm = 0.0f; int cnt = 0;
  #pragma unroll
  for (int t = 0; t < T; ++t) {
    float y = (out_row[(size_t)(t * NB + b) * V + v] - mean) * inv;
    vm += y;
    if (vm >= 1.0f) { cnt++; vm = 0.0f; }
  }
  float m = (float)cnt * 0.25f;
  out[(size_t)b * V + v] = m;
  float s = snn_block_reduce(m, red);
  if (threadIdx.x == 0) atomicAdd(&msum[b], s);
}

__global__ void snn_finalize2_kernel(float* __restrict__ out, const float* __restrict__ msum) {
  int b = blockIdx.x / (V / 256);
  int v = (blockIdx.x % (V / 256)) * 256 + threadIdx.x;
  out[(size_t)b * V + v] /= msum[b];
}

// ---------------------------------------------------------------------------
extern "C" void kernel_launch(void* const* d_in, const int* in_sizes, int n_in,
                              void* d_out, int out_size, void* d_ws, size_t ws_size,
                              hipStream_t stream) {
  (void)in_sizes; (void)n_in; (void)out_size; (void)ws_size;
  const float* x    = (const float*)d_in[0];
  const float* W    = (const float*)d_in[1];
  const float* bias = (const float*)d_in[2];
  float* out = (float*)d_out;
  char* ws = (char*)d_ws;

  _Float16* WH  = (_Float16*)(ws + WH_OFF);
  _Float16* S1H = (_Float16*)(ws + S1_OFF);
  float* rowb   = (float*)(ws + ROW_OFF);
  float* Cg     = (float*)(ws + C_OFF);
  float* Gg     = (float*)(ws + G_OFF);
  float* csW    = (float*)(ws + CSW_OFF);
  float* csWb   = (float*)(ws + CSWB_OFF);
  float* sumTS  = (float*)(ws + SUMTS_OFF);
  float* E2     = (float*)(ws + E2_OFF);
  float* msum   = (float*)(ws + MSUM_OFF);
  float* stats  = (float*)(ws + STATS_OFF);

  snn_zero_kernel<<<1024, 256, 0, stream>>>(Cg, ZERO_N);
  snn_convert_kernel<<<V / 32, 256, 0, stream>>>(W, WH, csW, csWb, bias);
  snn_if1_kernel<<<dim3(NB * E / 256, S / 64), 256, 0, stream>>>(x, S1H, sumTS);
  // C = W^T W : K=32000 split into 8 chunks -> 288 blocks of WMMA work
  snn_gram_kernel<false><<<dim3(6, 6, 8), 256, 0, stream>>>(WH, Cg, V, 8, 0);
  // G_b = S1_b^T S1_b : K=1024 (rows remapped through [T][NB][S] layout)
  snn_gram_kernel<true><<<dim3(6, 6, NB * 4), 256, 0, stream>>>(S1H, Gg, T * S, 4, NB * S);
  snn_reduceE2_kernel<<<dim3(E * E / 2048, NB), 256, 0, stream>>>(Gg, Cg, E2);
  snn_rowmat_kernel<<<V / 8, 256, 0, stream>>>(W, S1H, bias, rowb);
  snn_meanvar_kernel<<<1, 256, 0, stream>>>(sumTS, csW, csWb, bias, E2, stats);
  snn_finalize1_kernel<<<NB * V / 256, 256, 0, stream>>>(rowb, stats, out, msum);
  snn_finalize2_kernel<<<NB * V / 256, 256, 0, stream>>>(out, msum);
}
